// CausalSelfAttention_27513560498248
// MI455X (gfx1250) — compile-verified, matrix operand order checked by reading, not run on hardware
//
#include <hip/hip_runtime.h>
#include <hip/hip_bf16.h>

// ---- problem constants (reference: B=2, T=2048, L=1024, H=16, Dh=64) ----
#define BATCH     2
#define SEQ       2048
#define EMBED     1024
#define NUM_HEADS 16
#define HEAD_DIM  64

typedef __attribute__((ext_vector_type(16))) __bf16 v16bf;
typedef __attribute__((ext_vector_type(8)))  float  v8f;

// Async global->LDS (CDNA5): lowers to global_load_async_to_lds_b128
// (ASYNCcnt-tracked).  Probe-derived prototype:
//   void (int4 addrspace(1)* src, int4 addrspace(3)* dst, imm int off, imm int cpol)
#if defined(__has_builtin)
#if __has_builtin(__builtin_amdgcn_global_load_async_to_lds_b128)
#define ASYNC_LDS 1
#endif
#endif

typedef int v4i_vs __attribute__((vector_size(16)));
typedef __attribute__((address_space(1))) v4i_vs as1_v4i;
typedef __attribute__((address_space(3))) v4i_vs as3_v4i;

__device__ __forceinline__ void wait_async_lds() {
#if defined(ASYNC_LDS)
  asm volatile("s_wait_asynccnt 0x0" ::: "memory");
#endif
}

// ---- WMMA fragment helpers (CDNA5 16x16x32 bf16 layouts, ISA 7.12.2) ----
__device__ __forceinline__ int kbase(int vg, int half) {
  return (vg < 4 ? 2 * vg : 16 + 2 * (vg - 4)) + 8 * half;
}

// Fragment whose per-lane-owned dimension is the row of a row-major array
// (A-fragments; B-fragments of a transposed matrix): K pairs contiguous.
__device__ __forceinline__ v16bf load_frag_rowpair(const __bf16* __restrict__ base,
                                                   int row, int stride, int half, int koff) {
  v16bf f;
#pragma unroll
  for (int vg = 0; vg < 8; ++vg) {
    int k = kbase(vg, half) + koff;
    f[2 * vg]     = base[row * stride + k];
    f[2 * vg + 1] = base[row * stride + k + 1];
  }
  return f;
}

// B-fragment from a row-major B[k][n]: lane owns column n, gather down rows.
__device__ __forceinline__ v16bf load_frag_colgather(const __bf16* __restrict__ base,
                                                     int col, int stride, int half) {
  v16bf f;
#pragma unroll
  for (int vg = 0; vg < 8; ++vg) {
    int k = kbase(vg, half);
    f[2 * vg]     = base[k * stride + col];
    f[2 * vg + 1] = base[(k + 1) * stride + col];
  }
  return f;
}

__device__ __forceinline__ v8f wmma_bf16f32(v16bf a, v16bf b, v8f c) {
  return __builtin_amdgcn_wmma_f32_16x16x32_bf16(false, a, false, b, (short)0, c, false, false);
}

// ---- GEMM staging helpers (fp32 -> bf16 conversion in VALU) ----
__device__ __forceinline__ void stage_a_f32(const float* __restrict__ A, __bf16* __restrict__ As,
                                            int m0, int k0, int lda, int tid) {
#pragma unroll
  for (int i = 0; i < 16; ++i) {
    int e = tid + 256 * i;
    int m = e >> 5, k = e & 31;
    As[m * 34 + k] = (__bf16)A[(size_t)(m0 + m) * lda + k0 + k];
  }
}
__device__ __forceinline__ void stage_a_bf16(const __bf16* __restrict__ A, __bf16* __restrict__ As,
                                             int m0, int k0, int lda, int tid) {
#pragma unroll
  for (int i = 0; i < 16; ++i) {
    int e = tid + 256 * i;
    int m = e >> 5, k = e & 31;
    As[m * 34 + k] = A[(size_t)(m0 + m) * lda + k0 + k];
  }
}
__device__ __forceinline__ void stage_b_f32(const float* __restrict__ W, __bf16* __restrict__ Ws,
                                            int n0, int k0, int ldw, int tid) {
#pragma unroll
  for (int i = 0; i < 16; ++i) {
    int e = tid + 256 * i;
    int k = e >> 7, n = e & 127;
    Ws[n * 34 + k] = (__bf16)W[(size_t)(k0 + k) * ldw + n0 + n];
  }
}

// =====================================================================
// Kernel 1: qkv = x @ W_attn  -> scatter to Q/K/V [B,H,T,Dh] bf16.
// C is [4096 x 3072], K = 1024.  128x128 tile/block, BK=32, double-buffered.
// =====================================================================
__global__ __launch_bounds__(256) void qkv_gemm_kernel(const float* __restrict__ X,
                                                       const float* __restrict__ W,
                                                       __bf16* __restrict__ Q,
                                                       __bf16* __restrict__ K,
                                                       __bf16* __restrict__ V) {
  __shared__ __bf16 Xs[2][128 * 34];
  __shared__ __bf16 Ws[2][128 * 34];

  const int tid  = threadIdx.x;
  const int wave = tid >> 5;
  const int lane = tid & 31;
  const int half = lane >> 4;
  const int ln16 = lane & 15;
  const int n0 = blockIdx.x * 128;
  const int m0 = blockIdx.y * 128;

  v8f acc[8];
#pragma unroll
  for (int i = 0; i < 8; ++i) acc[i] = {};

  stage_a_f32(X, Xs[0], m0, 0, EMBED, tid);
  stage_b_f32(W, Ws[0], n0, 0, 3 * EMBED, tid);
  __syncthreads();

  int cur = 0;
  for (int k0 = 0; k0 < EMBED; k0 += 32) {
    if (k0 + 32 < EMBED) {             // prefetch next tile into other buffer
      stage_a_f32(X, Xs[cur ^ 1], m0, k0 + 32, EMBED, tid);
      stage_b_f32(W, Ws[cur ^ 1], n0, k0 + 32, 3 * EMBED, tid);
    }
    v16bf a = load_frag_rowpair(Xs[cur] + wave * 16 * 34, ln16, 34, half, 0);
#pragma unroll
    for (int t = 0; t < 8; ++t) {
      v16bf b = load_frag_rowpair(Ws[cur] + t * 16 * 34, ln16, 34, half, 0);
      acc[t] = wmma_bf16f32(a, b, acc[t]);
    }
    __syncthreads();
    cur ^= 1;
  }

#pragma unroll
  for (int t = 0; t < 8; ++t) {
#pragma unroll
    for (int r = 0; r < 8; ++r) {
      int row = m0 + wave * 16 + r + 8 * half;  // b*T + t index (0..4095)
      int col = n0 + t * 16 + ln16;             // 0..3071
      int seg = col >> 10;                      // 0=Q 1=K 2=V
      int l   = col & 1023;
      int h   = l >> 6;
      int d   = l & 63;
      int b   = row >> 11;
      int tq  = row & 2047;
      __bf16* dst = (seg == 0) ? Q : (seg == 1 ? K : V);
      dst[(((size_t)b * NUM_HEADS + h) * SEQ + tq) * HEAD_DIM + d] = (__bf16)acc[t][r];
    }
  }
}

// =====================================================================
// Kernel 2: causal flash attention.  Block = 4 waves = 64 query rows of
// one (b,h).  K/V tiles (32 keys x 64 dh bf16, 4KB each) are staged with
// async global->LDS b128 copies, double buffered so tile j+1 fills while
// tile j's WMMAs run.
// =====================================================================
__device__ __forceinline__ void stage_kv_async(__bf16* Kb, __bf16* Vb, int key0,
                                               __bf16* Ksb, __bf16* Vsb, int tid) {
#if defined(ASYNC_LDS)
#pragma unroll
  for (int i = 0; i < 2; ++i) {
    int off = (tid + 128 * i) * 8;  // 256 x 16B chunks per 4KB tile
    __builtin_amdgcn_global_load_async_to_lds_b128(
        (as1_v4i*)(Kb + (size_t)key0 * HEAD_DIM + off),
        (as3_v4i*)(Ksb + off), 0, 0);
    __builtin_amdgcn_global_load_async_to_lds_b128(
        (as1_v4i*)(Vb + (size_t)key0 * HEAD_DIM + off),
        (as3_v4i*)(Vsb + off), 0, 0);
  }
#else
  const unsigned int* ksrc = (const unsigned int*)(Kb + (size_t)key0 * HEAD_DIM);
  const unsigned int* vsrc = (const unsigned int*)(Vb + (size_t)key0 * HEAD_DIM);
  unsigned int* kdst = (unsigned int*)Ksb;
  unsigned int* vdst = (unsigned int*)Vsb;
#pragma unroll
  for (int i = 0; i < 8; ++i) {
    int e = tid + 128 * i;          // 1024 dwords per tile
    kdst[e] = ksrc[e];
    vdst[e] = vsrc[e];
  }
#endif
}

__global__ __launch_bounds__(128) void flash_attn_kernel(const __bf16* __restrict__ Q,
                                                         __bf16* __restrict__ K,
                                                         __bf16* __restrict__ V,
                                                         __bf16* __restrict__ Y) {
  __shared__ alignas(16) __bf16 Ks[2][32 * 64];   // [key][dh], contiguous for b128 async
  __shared__ alignas(16) __bf16 Vs[2][32 * 64];
  __shared__ __bf16 Ps[4 * 16 * 34];              // per-wave P transpose staging

  const int tid    = threadIdx.x;
  const int wave   = tid >> 5;
  const int lane   = tid & 31;
  const int half   = lane >> 4;
  const int ln16   = lane & 15;
  const int bh     = blockIdx.y;              // b*H + h
  const int q_tile = blockIdx.x;              // 64 queries / block
  const int q_base = q_tile * 64 + wave * 16; // wave's first query row

  const __bf16* Qb = Q + (size_t)bh * SEQ * HEAD_DIM;
  __bf16* Kb = K + (size_t)bh * SEQ * HEAD_DIM;
  __bf16* Vb = V + (size_t)bh * SEQ * HEAD_DIM;

  v16bf qa0 = load_frag_rowpair(Qb + q_base * HEAD_DIM, ln16, HEAD_DIM, half, 0);
  v16bf qa1 = load_frag_rowpair(Qb + q_base * HEAD_DIM, ln16, HEAD_DIM, half, 32);

  v8f o[4];
#pragma unroll
  for (int i = 0; i < 4; ++i) o[i] = {};
  float m_run[8], l_run[8];
#pragma unroll
  for (int r = 0; r < 8; ++r) { m_run[r] = -1e30f; l_run[r] = 0.f; }

  const float scale = 0.03125f;        // rsqrt(EMBED=1024), faithful to reference
  const int num_kt = q_tile * 2 + 2;   // key tiles of 32 covering causal range

  // prologue: fill buffer 0
  stage_kv_async(Kb, Vb, 0, Ks[0], Vs[0], tid);
  wait_async_lds();
  __syncthreads();

  int cur = 0;
  for (int j = 0; j < num_kt; ++j) {
    const int key0 = j * 32;
    if (j + 1 < num_kt)   // async-fill next tile while computing this one
      stage_kv_async(Kb, Vb, key0 + 32, Ks[cur ^ 1], Vs[cur ^ 1], tid);

    if (key0 <= q_base + 15) {   // tile intersects this wave's causal range
      const __bf16* Kt = Ks[cur];
      const __bf16* Vt = Vs[cur];
      // S = Q . K^T : two 16-key column tiles, K-dim = Dh = 64 (2 steps each)
      v8f s[2];
      s[0] = {}; s[1] = {};
#pragma unroll
      for (int nt = 0; nt < 2; ++nt) {
        v16bf kb0 = load_frag_rowpair(Kt + nt * 16 * 64, ln16, 64, half, 0);
        v16bf kb1 = load_frag_rowpair(Kt + nt * 16 * 64, ln16, 64, half, 32);
        s[nt] = wmma_bf16f32(qa0, kb0, s[nt]);
        s[nt] = wmma_bf16f32(qa1, kb1, s[nt]);
      }

      // causal mask + scale (lane owns column n = ln16, row = r + 8*half)
#pragma unroll
      for (int r = 0; r < 8; ++r) {
        int qg = q_base + r + 8 * half;
#pragma unroll
        for (int nt = 0; nt < 2; ++nt) {
          int key = key0 + nt * 16 + ln16;
          float sv = s[nt][r];
          s[nt][r] = (key <= qg) ? sv * scale : -1e30f;
        }
      }

      // online softmax: per-row reductions stay within 16-lane halves
      __bf16* Pw = Ps + wave * 16 * 34;
#pragma unroll
      for (int r = 0; r < 8; ++r) {
        float t = fmaxf(s[0][r], s[1][r]);
#pragma unroll
        for (int mk = 1; mk < 16; mk <<= 1) t = fmaxf(t, __shfl_xor(t, mk, 32));
        float m_new = fmaxf(m_run[r], t);
        float alpha = __expf(m_run[r] - m_new);
        float p0 = __expf(s[0][r] - m_new);
        float p1 = __expf(s[1][r] - m_new);
        float rs = p0 + p1;
#pragma unroll
        for (int mk = 1; mk < 16; mk <<= 1) rs += __shfl_xor(rs, mk, 32);
        l_run[r] = l_run[r] * alpha + rs;
        m_run[r] = m_new;
#pragma unroll
        for (int t4 = 0; t4 < 4; ++t4) o[t4][r] *= alpha;
        int mrow = r + 8 * half;
        Pw[mrow * 34 + ln16]      = (__bf16)p0;
        Pw[mrow * 34 + 16 + ln16] = (__bf16)p1;
      }

      // intra-wave LDS RAW: drain DS stores before the fragment gather
      asm volatile("s_wait_dscnt 0x0" ::: "memory");

      v16bf pf = load_frag_rowpair(Pw, ln16, 34, half, 0);  // P as A (16x32)
#pragma unroll
      for (int t4 = 0; t4 < 4; ++t4) {
        v16bf vb = load_frag_colgather(Vt + t4 * 16, ln16, 64, half);
        o[t4] = wmma_bf16f32(pf, vb, o[t4]);
      }
    }

    wait_async_lds();   // next tile's async copies complete during compute
    __syncthreads();
    cur ^= 1;
  }

  // epilogue: Y[b, q, h*Dh + d] = O / l   (bf16 feed for projection GEMM)
  const int b = bh >> 4;
  const int h = bh & 15;
#pragma unroll
  for (int t4 = 0; t4 < 4; ++t4) {
#pragma unroll
    for (int r = 0; r < 8; ++r) {
      int qg = q_base + r + 8 * half;
      int d  = t4 * 16 + ln16;
      float val = o[t4][r] / l_run[r];
      Y[((size_t)(b * SEQ + qg)) * EMBED + h * HEAD_DIM + d] = (__bf16)val;
    }
  }
}

// =====================================================================
// Kernel 3: out = Y @ W_proj   (4096x1024, K=1024), fp32 out, double-buffered.
// =====================================================================
__global__ __launch_bounds__(256) void proj_gemm_kernel(const __bf16* __restrict__ Y,
                                                        const float* __restrict__ W,
                                                        float* __restrict__ Out) {
  __shared__ __bf16 Ys[2][128 * 34];
  __shared__ __bf16 Ws[2][128 * 34];

  const int tid  = threadIdx.x;
  const int wave = tid >> 5;
  const int lane = tid & 31;
  const int half = lane >> 4;
  const int ln16 = lane & 15;
  const int n0 = blockIdx.x * 128;
  const int m0 = blockIdx.y * 128;

  v8f acc[8];
#pragma unroll
  for (int i = 0; i < 8; ++i) acc[i] = {};

  stage_a_bf16(Y, Ys[0], m0, 0, EMBED, tid);
  stage_b_f32(W, Ws[0], n0, 0, EMBED, tid);
  __syncthreads();

  int cur = 0;
  for (int k0 = 0; k0 < EMBED; k0 += 32) {
    if (k0 + 32 < EMBED) {
      stage_a_bf16(Y, Ys[cur ^ 1], m0, k0 + 32, EMBED, tid);
      stage_b_f32(W, Ws[cur ^ 1], n0, k0 + 32, EMBED, tid);
    }
    v16bf a = load_frag_rowpair(Ys[cur] + wave * 16 * 34, ln16, 34, half, 0);
#pragma unroll
    for (int t = 0; t < 8; ++t) {
      v16bf b = load_frag_rowpair(Ws[cur] + t * 16 * 34, ln16, 34, half, 0);
      acc[t] = wmma_bf16f32(a, b, acc[t]);
    }
    __syncthreads();
    cur ^= 1;
  }

#pragma unroll
  for (int t = 0; t < 8; ++t) {
#pragma unroll
    for (int r = 0; r < 8; ++r) {
      int row = m0 + wave * 16 + r + 8 * half;
      int col = n0 + t * 16 + ln16;
      Out[(size_t)row * EMBED + col] = acc[t][r];
    }
  }
}

extern "C" void kernel_launch(void* const* d_in, const int* in_sizes, int n_in,
                              void* d_out, int out_size, void* d_ws, size_t ws_size,
                              hipStream_t stream) {
  const float* x      = (const float*)d_in[0];   // [2, 2048, 1024]
  const float* w_attn = (const float*)d_in[1];   // [1024, 3072]
  const float* w_proj = (const float*)d_in[2];   // [1024, 1024]
  float* out = (float*)d_out;                    // [2, 2048, 1024]

  // workspace layout: Q | K | V | Y, each B*T*L bf16 elements (8 MiB each)
  const size_t elems = (size_t)BATCH * SEQ * EMBED;
  __bf16* qw = (__bf16*)d_ws;
  __bf16* kw = qw + elems;
  __bf16* vw = kw + elems;
  __bf16* yw = vw + elems;

  qkv_gemm_kernel<<<dim3(3 * EMBED / 128, BATCH * SEQ / 128), 256, 0, stream>>>(
      x, w_attn, qw, kw, vw);

  flash_attn_kernel<<<dim3(SEQ / 64, BATCH * NUM_HEADS), 128, 0, stream>>>(
      qw, kw, vw, yw);

  proj_gemm_kernel<<<dim3(EMBED / 128, BATCH * SEQ / 128), 256, 0, stream>>>(
      yw, w_proj, out);
}
